// ProbSparseAttention_63685775065576
// MI455X (gfx1250) — matrix operand order repaired and checked by reading, not compiled, not run on hardware
//
#include <hip/hip_runtime.h>

// ===========================================================================
// ProbSparse attention for MI455X (gfx1250, wave32, WMMA).
// Pipeline:
//  0) cvt_bf16_kernel x7  : fp32 -> bf16 for inputs and weight matrices.
//  1) gemm_bf16_kernel x3 : Q/K/V projections (bf16 WMMA 16x16x32, f32 acc),
//     async double-buffered LDS staging, output in (B,H,L,64) bf16 layout.
//  2) score_kernel        : M[b,h,l] = max_s - mean_s over 40 sampled keys.
//  3) topk_kernel         : iterative top-40 argmax per (b,h).
//  4) zero4_kernel        : zero the sparse `full` buffer (bf16).
//  5) attn_kernel         : fused flash attention for the 40 selected rows,
//     WMMA for QK^T and P*V, online softmax, scatter bf16 into `full`.
//  6) gemm_bf16_kernel    : out = full @ Wo^T + bo (fp32 output).
// ===========================================================================

typedef __bf16 bf16_t;
typedef __attribute__((ext_vector_type(16))) __bf16 v16bf;
typedef __attribute__((ext_vector_type(8)))  __bf16 v8bf;
typedef __attribute__((ext_vector_type(8)))  float  v8f;
typedef __attribute__((ext_vector_type(4)))  float  v4f;
typedef int v4i_t __attribute__((vector_size(4 * sizeof(int))));

#define B_    4
#define H_    16
#define L_    4096
#define DM    1024
#define DK    64
#define NTOP  40
#define NSAMP 40
#define SCALE 0.125f

// CDNA5 async copy path (GLOBAL_LOAD_ASYNC_TO_LDS_B128 + ASYNCcnt), guarded
// so the file still compiles if this toolchain doesn't declare the builtins.
#if __has_builtin(__builtin_amdgcn_global_load_async_to_lds_b128) && \
    __has_builtin(__builtin_amdgcn_s_wait_asynccnt)
#define USE_ASYNC_COPY 1
#else
#define USE_ASYNC_COPY 0
#endif

static __device__ __forceinline__ void wait_async_all() {
#if USE_ASYNC_COPY
  __builtin_amdgcn_s_wait_asynccnt(0);
#endif
}

static __device__ __forceinline__ v8f vzero8() {
  v8f z;
#pragma unroll
  for (int i = 0; i < 8; ++i) z[i] = 0.0f;
  return z;
}

static __device__ __forceinline__ v8f wmma_bf16(v16bf a, v16bf b, v8f c) {
  // D = A(16x32 bf16) * B(32x16 bf16) + C(16x16 f32)
  return __builtin_amdgcn_wmma_f32_16x16x32_bf16(
      /*neg_a=*/false, a, /*neg_b=*/false, b,
      /*c_mod=*/(short)0, c, /*reuse_a=*/false, /*reuse_b=*/false);
}

static __device__ __forceinline__ v16bf cat16(v8bf lo, v8bf hi) {
  return __builtin_shufflevector(lo, hi, 0, 1, 2, 3, 4, 5, 6, 7,
                                 8, 9, 10, 11, 12, 13, 14, 15);
}

// A-matrix fragment (16x32, 16-bit): lane l -> row M = l&15;
// elems 0..7 -> K = koff..koff+7, elems 8..15 -> K = 16+koff..23+koff,
// koff = 8*(l>=16).  Memory row-major, K contiguous, pitch in elems.
static __device__ __forceinline__ v16bf load_a_frag(const bf16_t* base,
                                                    int pitch, int lane) {
  const bf16_t* p = base + (lane & 15) * pitch + ((lane & 16) ? 8 : 0);
  return cat16(*(const v8bf*)p, *(const v8bf*)(p + 16));
}

// B-matrix fragment (32x16, 16-bit): lane l -> col N = l&15;
// elems 0..7 -> K = koff..koff+7, elems 8..15 -> K = 16+koff..23+koff,
// koff = 8*(l>=16).  `base` row-major with N as the row and K
// contiguous within the row (i.e. B[k][n] = base[n*pitch + k]).
static __device__ __forceinline__ v16bf load_b_frag(const bf16_t* base,
                                                    int pitch, int lane) {
  const bf16_t* p = base + (lane & 15) * pitch + ((lane & 16) ? 8 : 0);
  return cat16(*(const v8bf*)p, *(const v8bf*)(p + 16));
}

static __device__ __forceinline__ v8f half_shfl_max(v8f v) {
#pragma unroll
  for (int mk = 1; mk < 16; mk <<= 1) {
#pragma unroll
    for (int r = 0; r < 8; ++r) v[r] = fmaxf(v[r], __shfl_xor(v[r], mk, 32));
  }
  return v;
}

static __device__ __forceinline__ v8f half_shfl_sum(v8f v) {
#pragma unroll
  for (int mk = 1; mk < 16; mk <<= 1) {
#pragma unroll
    for (int r = 0; r < 8; ++r) v[r] += __shfl_xor(v[r], mk, 32);
  }
  return v;
}

// ===========================================================================
// fp32 -> bf16 elementwise (8 elems/thread, b128 in / b128 out)
// ===========================================================================
__global__ __launch_bounds__(256) void cvt_bf16_kernel(
    const float* __restrict__ src, bf16_t* __restrict__ dst) {
  size_t i = ((size_t)blockIdx.x * 256 + threadIdx.x) * 8;
  v4f a = *(const v4f*)(src + i);
  v4f b = *(const v4f*)(src + i + 4);
  v8bf o;
#pragma unroll
  for (int e = 0; e < 4; ++e) {
    o[e] = (bf16_t)a[e];
    o[4 + e] = (bf16_t)b[e];
  }
  *(v8bf*)(dst + i) = o;
}

__global__ void zero4_kernel(float* __restrict__ p) {
  size_t i = ((size_t)blockIdx.x * blockDim.x + threadIdx.x) * 4;
  v4f z;
  z[0] = z[1] = z[2] = z[3] = 0.0f;
  *(v4f*)(p + i) = z;
}

// ===========================================================================
// GEMM: Y[m,n] = sum_k A[m,k] * W[n,k] + bias[n]
// A: (M=16384,1024) bf16 row-major, W: (1024,1024) bf16 row-major.
// mode 0: fp32 out, row-major (m,n).  mode 1: bf16 out, (B,H,L,64) layout.
// Block tile 128x128, k-chunk 32; 8 waves as 4(M)x2(N), wave tile 32x64.
// Double-buffered LDS staging via CDNA5 async copies (fallback: sync).
// ===========================================================================
static __device__ __forceinline__ void stage_pair(
    const bf16_t* gA, const bf16_t* gW, bf16_t (*Asb)[32], bf16_t (*Bsb)[32],
    int tid) {
#pragma unroll
  for (int i = 0; i < 2; ++i) {
    int f = tid + i * 256;     // b128 index 0..511
    int row = f >> 2;          // 0..127
    int c8 = (f & 3) << 3;     // 0,8,16,24 (bf16 elems)
#if USE_ASYNC_COPY
    __builtin_amdgcn_global_load_async_to_lds_b128(
        (v4i_t*)(gA + (size_t)row * DM + c8), (v4i_t*)(&Asb[row][c8]), 0, 0);
    __builtin_amdgcn_global_load_async_to_lds_b128(
        (v4i_t*)(gW + (size_t)row * DM + c8), (v4i_t*)(&Bsb[row][c8]), 0, 0);
#else
    v8bf va = *(const v8bf*)(gA + (size_t)row * DM + c8);
    v8bf vb = *(const v8bf*)(gW + (size_t)row * DM + c8);
    *(v8bf*)(&Asb[row][c8]) = va;
    *(v8bf*)(&Bsb[row][c8]) = vb;
#endif
  }
}

__global__ __launch_bounds__(256) void gemm_bf16_kernel(
    const bf16_t* __restrict__ A, const bf16_t* __restrict__ W,
    const float* __restrict__ bias, void* __restrict__ out, int mode) {
  __shared__ bf16_t As[2][128][32];  // 16 KB
  __shared__ bf16_t Bs[2][128][32];  // 16 KB

  const int tid = threadIdx.x;
  const int lane = tid & 31;
  const int wave = tid >> 5;
  const int wm = wave & 3;   // M sub-block (32 rows)
  const int wn = wave >> 2;  // N sub-block (64 cols)
  const int mBase = blockIdx.y * 128;
  const int nBase = blockIdx.x * 128;
  const bf16_t* gA = A + (size_t)mBase * DM;
  const bf16_t* gW = W + (size_t)nBase * DM;

  v8f acc[2][4];
#pragma unroll
  for (int mt = 0; mt < 2; ++mt)
#pragma unroll
    for (int nt = 0; nt < 4; ++nt) acc[mt][nt] = vzero8();

  // Prologue: stage k-chunk 0 into buffer 0.
  stage_pair(gA, gW, As[0], Bs[0], tid);
  wait_async_all();
  __syncthreads();

  for (int k0 = 0; k0 < DM; k0 += 32) {
    const int cur = (k0 >> 5) & 1;
    // Issue async copies for the next k-chunk while computing this one.
    if (k0 + 32 < DM)
      stage_pair(gA + k0 + 32, gW + k0 + 32, As[cur ^ 1], Bs[cur ^ 1], tid);

    v16bf af[2], bfr[4];
#pragma unroll
    for (int mt = 0; mt < 2; ++mt)
      af[mt] = load_a_frag(&As[cur][wm * 32 + mt * 16][0], 32, lane);
#pragma unroll
    for (int nt = 0; nt < 4; ++nt)
      bfr[nt] = load_b_frag(&Bs[cur][wn * 64 + nt * 16][0], 32, lane);

#pragma unroll
    for (int mt = 0; mt < 2; ++mt)
#pragma unroll
      for (int nt = 0; nt < 4; ++nt)
        acc[mt][nt] = wmma_bf16(af[mt], bfr[nt], acc[mt][nt]);

    wait_async_all();  // next tile landed (no-op on last iteration)
    __syncthreads();
  }

  // C/D layout: VGPR r, lanes 0-15 -> (M=r, N=lane); lanes 16-31 -> (M=8+r).
#pragma unroll
  for (int mt = 0; mt < 2; ++mt)
#pragma unroll
    for (int nt = 0; nt < 4; ++nt)
#pragma unroll
      for (int r = 0; r < 8; ++r) {
        int m = mBase + wm * 32 + mt * 16 + r + ((lane & 16) ? 8 : 0);
        int n = nBase + wn * 64 + nt * 16 + (lane & 15);
        float v = acc[mt][nt][r] + bias[n];
        if (mode == 0) {
          ((float*)out)[(size_t)m * DM + n] = v;
        } else {
          int b = m >> 12, t = m & (L_ - 1);
          int h = n >> 6, d = n & (DK - 1);
          ((bf16_t*)out)[(((size_t)(b * H_ + h)) * L_ + t) * DK + d] =
              (bf16_t)v;
        }
      }
}

// ===========================================================================
// Scoring: M[b,h,l] = max_s(QK_s) - mean_s(QK_s) over 40 sampled keys.
// ===========================================================================
__global__ __launch_bounds__(256) void score_kernel(
    const bf16_t* __restrict__ Qb, const bf16_t* __restrict__ Kb,
    const int* __restrict__ idx, float* __restrict__ Ms) {
  __shared__ bf16_t Ks[NSAMP][DK];  // 5 KB
  const int b = blockIdx.z, h = blockIdx.y;
  const size_t bh = (size_t)b * H_ + h;

  for (int e = threadIdx.x; e < NSAMP * DK; e += 256) {
    int s = e >> 6, d = e & (DK - 1);
    Ks[s][d] = Kb[(bh * L_ + idx[s]) * DK + d];
  }
  __syncthreads();

  const int l = blockIdx.x * 256 + threadIdx.x;
  const bf16_t* qp = Qb + (bh * L_ + l) * DK;
  float q[DK];
#pragma unroll
  for (int d8 = 0; d8 < 8; ++d8) {
    v8bf v = *(const v8bf*)(qp + d8 * 8);
#pragma unroll
    for (int e = 0; e < 8; ++e) q[d8 * 8 + e] = (float)v[e];
  }

  float mx = -3.0e38f, sum = 0.0f;
  for (int s = 0; s < NSAMP; ++s) {
    float acc = 0.0f;
#pragma unroll
    for (int d8 = 0; d8 < 8; ++d8) {
      v8bf v = *(const v8bf*)(&Ks[s][d8 * 8]);
#pragma unroll
      for (int e = 0; e < 8; ++e) acc += q[d8 * 8 + e] * (float)v[e];
    }
    mx = fmaxf(mx, acc);
    sum += acc;
  }
  Ms[bh * L_ + l] = mx - sum * (1.0f / (float)NSAMP);
}

// ===========================================================================
// Top-40 per (b,h): iterative parallel argmax, ties to the smaller index.
// ===========================================================================
__global__ __launch_bounds__(256) void topk_kernel(const float* __restrict__ Ms,
                                                   int* __restrict__ Mtop) {
  __shared__ float vals[L_];  // 16 KB
  __shared__ float rv[256];
  __shared__ int ri[256];
  const size_t bh = blockIdx.x;
  const int tid = threadIdx.x;

  for (int i = tid; i < L_; i += 256) vals[i] = Ms[bh * L_ + i];
  __syncthreads();

  for (int it = 0; it < NTOP; ++it) {
    float bv = -3.0e38f;
    int bi = 0x7fffffff;
    for (int i = tid; i < L_; i += 256) {
      float v = vals[i];
      if (v > bv || (v == bv && i < bi)) { bv = v; bi = i; }
    }
    rv[tid] = bv;
    ri[tid] = bi;
    __syncthreads();
    for (int off = 128; off > 0; off >>= 1) {
      if (tid < off) {
        float v = rv[tid + off];
        int j = ri[tid + off];
        if (v > rv[tid] || (v == rv[tid] && j < ri[tid])) {
          rv[tid] = v;
          ri[tid] = j;
        }
      }
      __syncthreads();
    }
    if (tid == 0) {
      Mtop[bh * NTOP + it] = ri[0];
      vals[ri[0]] = -3.0e38f;
    }
    __syncthreads();
  }
}

// ===========================================================================
// Fused sparse attention per (b,h): 40 rows padded to 48 (3 WMMA M-tiles).
// 4 waves flash-split S in 32-wide chunks; per-wave 48x64 f32 accumulators +
// online softmax stats in fragments; LDS merge; bf16 scatter into `full`.
// ===========================================================================
__global__ __launch_bounds__(128) void attn_kernel(
    const bf16_t* __restrict__ Qb, const bf16_t* __restrict__ Kb,
    const bf16_t* __restrict__ Vb, const int* __restrict__ Mtop,
    bf16_t* __restrict__ fullb) {
  // Loop phase: qred(6144) + vt(16384) + pb(4096) = 26624 B;
  // merge phase reuses the arena as Osh (4*48*64*4 = 49152 B).
  __shared__ __align__(16) char arena[4 * 48 * 64 * 4];
  __shared__ float msh[4][48], lsh[4][48];
  float(*Osh)[48][64] = (float(*)[48][64])arena;
  bf16_t* qred = (bf16_t*)arena;                 // [48][64]
  bf16_t* vt = (bf16_t*)(arena + 6144);          // [4][64][32] (d-major)
  bf16_t* pb = (bf16_t*)(arena + 6144 + 16384);  // [4][16][32]

  const int tid = threadIdx.x, lane = tid & 31, wave = tid >> 5;
  const int bh = blockIdx.x, b = bh >> 4, h = bh & 15;
  const bf16_t* Qbh = Qb + (size_t)bh * L_ * DK;
  const bf16_t* Kbh = Kb + (size_t)bh * L_ * DK;
  const bf16_t* Vbh = Vb + (size_t)bh * L_ * DK;

  for (int e = tid; e < 48 * DK; e += 128) {
    int r = e >> 6, d = e & (DK - 1);
    bf16_t v = (bf16_t)0.0f;
    if (r < NTOP) v = Qbh[(size_t)Mtop[bh * NTOP + r] * DK + d];
    qred[e] = v;
  }
  __syncthreads();

  v8f O[3][4], mrow[3], lrow[3];
#pragma unroll
  for (int mt = 0; mt < 3; ++mt) {
#pragma unroll
    for (int nt = 0; nt < 4; ++nt) O[mt][nt] = vzero8();
    v8f mi, li;
#pragma unroll
    for (int r = 0; r < 8; ++r) { mi[r] = -3.0e38f; li[r] = 0.0f; }
    mrow[mt] = mi;
    lrow[mt] = li;
  }

  bf16_t* vtw = vt + wave * 64 * 32;
  bf16_t* pbw = pb + wave * 16 * 32;

  for (int c = wave; c < L_ / 32; c += 4) {
    const int s0 = c * 32;
    // Prefetch next chunk's K/V rows (global_prefetch_b8).
    if (c + 4 < L_ / 32) {
      __builtin_prefetch(Kbh + (size_t)(s0 + 128 + lane) * DK, 0, 1);
      __builtin_prefetch(Vbh + (size_t)(s0 + 128 + lane) * DK, 0, 1);
    }

    // Stage V chunk transposed (vtw[d][s]) for contiguous P*V B-fragments.
    {
      const bf16_t* vrow = Vbh + (size_t)(s0 + lane) * DK;
#pragma unroll
      for (int d8 = 0; d8 < 8; ++d8) {
        v8bf v = *(const v8bf*)(vrow + d8 * 8);
#pragma unroll
        for (int e = 0; e < 8; ++e) vtw[(d8 * 8 + e) * 32 + lane] = v[e];
      }
    }
    // K B-fragments straight from global: B[k=d][n=s] = K[s][d].
    v16bf bk[2][2];
#pragma unroll
    for (int st = 0; st < 2; ++st)
#pragma unroll
      for (int ks = 0; ks < 2; ++ks)
        bk[st][ks] =
            load_b_frag(Kbh + (size_t)(s0 + st * 16) * DK + ks * 32, DK, lane);
    __syncthreads();

    v16bf bv[4];
#pragma unroll
    for (int nt = 0; nt < 4; ++nt)
      bv[nt] = load_b_frag(vtw + nt * 16 * 32, 32, lane);

#pragma unroll
    for (int mt = 0; mt < 3; ++mt) {
      v16bf a0 = load_a_frag(qred + mt * 16 * DK, DK, lane);
      v16bf a1 = load_a_frag(qred + mt * 16 * DK + 32, DK, lane);

      v8f S[2];
#pragma unroll
      for (int st = 0; st < 2; ++st) {
        v8f z = vzero8();
        z = wmma_bf16(a0, bk[st][0], z);
        z = wmma_bf16(a1, bk[st][1], z);
#pragma unroll
        for (int r = 0; r < 8; ++r) z[r] *= SCALE;
        S[st] = z;
      }

      // Online softmax (rows live on elem r within 16-lane halves).
      v8f t;
#pragma unroll
      for (int r = 0; r < 8; ++r) t[r] = fmaxf(S[0][r], S[1][r]);
      t = half_shfl_max(t);
      v8f mnew, alpha;
#pragma unroll
      for (int r = 0; r < 8; ++r) {
        mnew[r] = fmaxf(mrow[mt][r], t[r]);
        alpha[r] = __expf(mrow[mt][r] - mnew[r]);
      }
      mrow[mt] = mnew;
#pragma unroll
      for (int st = 0; st < 2; ++st)
#pragma unroll
        for (int r = 0; r < 8; ++r) S[st][r] = __expf(S[st][r] - mnew[r]);
      v8f rs;
#pragma unroll
      for (int r = 0; r < 8; ++r) rs[r] = S[0][r] + S[1][r];
      rs = half_shfl_sum(rs);
#pragma unroll
      for (int r = 0; r < 8; ++r)
        lrow[mt][r] = lrow[mt][r] * alpha[r] + rs[r];
#pragma unroll
      for (int nt = 0; nt < 4; ++nt)
#pragma unroll
        for (int r = 0; r < 8; ++r) O[mt][nt][r] *= alpha[r];

      // Re-stage P (C-layout -> row-major) so it reloads as an A-fragment.
      const int rofs = (lane & 16) ? 8 : 0;
#pragma unroll
      for (int st = 0; st < 2; ++st)
#pragma unroll
        for (int r = 0; r < 8; ++r)
          pbw[(r + rofs) * 32 + st * 16 + (lane & 15)] = (bf16_t)S[st][r];
      __syncthreads();
      v16bf pa = load_a_frag(pbw, 32, lane);
#pragma unroll
      for (int nt = 0; nt < 4; ++nt)
        O[mt][nt] = wmma_bf16(pa, bv[nt], O[mt][nt]);
    }
  }
  __syncthreads();  // all waves done with qred/vt/pb before Osh overwrite

#pragma unroll
  for (int mt = 0; mt < 3; ++mt) {
    const int rofs = (lane & 16) ? 8 : 0;
    if ((lane & 15) == 0) {
#pragma unroll
      for (int r = 0; r < 8; ++r) {
        msh[wave][mt * 16 + rofs + r] = mrow[mt][r];
        lsh[wave][mt * 16 + rofs + r] = lrow[mt][r];
      }
    }
#pragma unroll
    for (int nt = 0; nt < 4; ++nt)
#pragma unroll
      for (int r = 0; r < 8; ++r)
        Osh[wave][mt * 16 + rofs + r][nt * 16 + (lane & 15)] = O[mt][nt][r];
  }
  __syncthreads();

  for (int e = tid; e < NTOP * DK; e += 128) {
    int r = e >> 6, d = e & (DK - 1);
    float mm = -3.0e38f;
#pragma unroll
    for (int w = 0; w < 4; ++w) mm = fmaxf(mm, msh[w][r]);
    float Ls = 0.0f, val = 0.0f;
#pragma unroll
    for (int w = 0; w < 4; ++w) {
      float f = __expf(msh[w][r] - mm);
      Ls += lsh[w][r] * f;
      val += Osh[w][r][d] * f;
    }
    int tpos = Mtop[bh * NTOP + r];
    fullb[((size_t)b * L_ + tpos) * DM + h * DK + d] = (bf16_t)(val / Ls);
  }
}

// ===========================================================================
extern "C" void kernel_launch(void* const* d_in, const int* in_sizes, int n_in,
                              void* d_out, int out_size, void* d_ws,
                              size_t ws_size, hipStream_t stream) {
  (void)in_sizes; (void)n_in; (void)out_size; (void)ws_size;
  const float* queries = (const float*)d_in[0];
  const float* keys = (const float*)d_in[1];
  const float* values = (const float*)d_in[2];
  const float* Wq = (const float*)d_in[3];
  const float* bq = (const float*)d_in[4];
  const float* Wk = (const float*)d_in[5];
  const float* bk = (const float*)d_in[6];
  const float* Wv = (const float*)d_in[7];
  const float* bv = (const float*)d_in[8];
  const float* Wo = (const float*)d_in[9];
  const float* bo = (const float*)d_in[10];
  const int* idxs = (const int*)d_in[11];

  char* ws = (char*)d_ws;
  const size_t BIG = (size_t)B_ * H_ * L_ * DK * sizeof(bf16_t);  // 32 MB
  const size_t WB = (size_t)DM * DM * sizeof(bf16_t);             // 2 MB
  bf16_t* Qb = (bf16_t*)(ws + 0 * BIG);
  bf16_t* Kb = (bf16_t*)(ws + 1 * BIG);
  bf16_t* Vb = (bf16_t*)(ws + 2 * BIG);
  bf16_t* Xq = (bf16_t*)(ws + 3 * BIG);  // later reused as `full` (bf16)
  bf16_t* Xk = (bf16_t*)(ws + 4 * BIG);
  bf16_t* Xv = (bf16_t*)(ws + 5 * BIG);
  bf16_t* Wqb = (bf16_t*)(ws + 6 * BIG);
  bf16_t* Wkb = (bf16_t*)(ws + 6 * BIG + 1 * WB);
  bf16_t* Wvb = (bf16_t*)(ws + 6 * BIG + 2 * WB);
  bf16_t* Wob = (bf16_t*)(ws + 6 * BIG + 3 * WB);
  float* Ms = (float*)(ws + 6 * BIG + 4 * WB);
  int* Mtop = (int*)(ws + 6 * BIG + 4 * WB + (size_t)B_ * H_ * L_ * 4);
  bf16_t* fullb = Xq;

  const int NBIG = B_ * L_ * DM;  // 16,777,216 elems
  cvt_bf16_kernel<<<NBIG / 8 / 256, 256, 0, stream>>>(queries, Xq);
  cvt_bf16_kernel<<<NBIG / 8 / 256, 256, 0, stream>>>(keys, Xk);
  cvt_bf16_kernel<<<NBIG / 8 / 256, 256, 0, stream>>>(values, Xv);
  cvt_bf16_kernel<<<DM * DM / 8 / 256, 256, 0, stream>>>(Wq, Wqb);
  cvt_bf16_kernel<<<DM * DM / 8 / 256, 256, 0, stream>>>(Wk, Wkb);
  cvt_bf16_kernel<<<DM * DM / 8 / 256, 256, 0, stream>>>(Wv, Wvb);
  cvt_bf16_kernel<<<DM * DM / 8 / 256, 256, 0, stream>>>(Wo, Wob);

  dim3 gG(DM / 128, (B_ * L_) / 128);  // (8, 128)
  gemm_bf16_kernel<<<gG, 256, 0, stream>>>(Xq, Wqb, bq, Qb, 1);
  gemm_bf16_kernel<<<gG, 256, 0, stream>>>(Xk, Wkb, bk, Kb, 1);
  gemm_bf16_kernel<<<gG, 256, 0, stream>>>(Xv, Wvb, bv, Vb, 1);

  score_kernel<<<dim3(L_ / 256, H_, B_), 256, 0, stream>>>(Qb, Kb, idxs, Ms);
  topk_kernel<<<B_ * H_, 256, 0, stream>>>(Ms, Mtop);

  // fullb is bf16: zero its bytes (NBIG bf16 elems == NBIG/2 floats).
  zero4_kernel<<<NBIG / 2 / 4 / 256, 256, 0, stream>>>((float*)fullb);
  attn_kernel<<<B_ * H_, 128, 0, stream>>>(Qb, Kb, Vb, Mtop, fullb);

  gemm_bf16_kernel<<<gG, 256, 0, stream>>>(fullb, Wob, bo, (float*)d_out, 0);
}
